// RegionLoss_3D_info_61263413510188
// MI455X (gfx1250) — matrix operand order repaired and checked by reading, not compiled, not run on hardware
//
#include <hip/hip_runtime.h>
#include <math.h>

typedef __attribute__((ext_vector_type(16))) _Float16 v16h;
typedef __attribute__((ext_vector_type(8)))  _Float16 v8h;
typedef __attribute__((ext_vector_type(8)))  float    v8f;

// Problem constants (from reference): B=8, C=16, DHW=64, S=8
#define NROWS   8192          // 2 * B * S^3
#define HALF_N  4096
#define INV_T   10.0f         // 1 / TEMPERATURE
#define MAXL    10.0f         // fixed logit max (rows are unit-norm -> l <= 1/T)

// ---------------------------------------------------------------------------
// Kernel 1: 8x8x8 average pool of both volumes into mm[8192][16].
// One thread per pooled output; inner float4 loads give 256B-contiguous
// requests per 8-lane group (full 64-float W rows). HBM-bound: 268MB reads.
// ---------------------------------------------------------------------------
__global__ void __launch_bounds__(256)
pool_kernel(const float* __restrict__ p1, const float* __restrict__ p2,
            float* __restrict__ mm) {
  int gid = blockIdx.x * 256 + threadIdx.x;   // 131072 total
  int t    = gid >> 16;                       // which tensor (0=p1,1=p2)
  int rem  = gid & 65535;
  int b    = rem >> 13;
  int rem2 = rem & 8191;
  int c    = rem2 >> 9;
  int s    = rem2 & 511;                      // spatial index zi*64+yi*8+xi
  int zi = s >> 6, yi = (s >> 3) & 7, xi = s & 7;

  const float* __restrict__ p = t ? p2 : p1;
  size_t base = ((((size_t)(b * 16 + c)) * 64 + (size_t)zi * 8) * 64
                 + (size_t)yi * 8) * 64 + (size_t)xi * 8;
  float sum = 0.0f;
#pragma unroll 2
  for (int dz = 0; dz < 8; ++dz) {
#pragma unroll
    for (int dy = 0; dy < 8; ++dy) {
      const float* row = p + base + (size_t)dz * 4096 + (size_t)dy * 64;
      float4 a = *(const float4*)(row);
      float4 q = *(const float4*)(row + 4);
      sum += a.x + a.y + a.z + a.w + q.x + q.y + q.z + q.w;
    }
  }
  int rowi = t * HALF_N + b * 512 + s;        // row of mm
  mm[rowi * 16 + c] = sum * (1.0f / 512.0f);
}

// ---------------------------------------------------------------------------
// Kernel 2: per-row Linear->ReLU->Linear->L2 normalize (fp32, 8M FLOP total),
// then split each f value into f16 hi + f16 lo for the WMMA pass.
// ---------------------------------------------------------------------------
__global__ void __launch_bounds__(256)
mlp_kernel(const float* __restrict__ mm,
           const float* __restrict__ w1, const float* __restrict__ b1,
           const float* __restrict__ w2, const float* __restrict__ b2,
           float* __restrict__ fout,
           _Float16* __restrict__ fhi, _Float16* __restrict__ flo) {
  __shared__ float sw1[256], sw2[256], sb1[16], sb2[16];
  int tid = threadIdx.x;
  sw1[tid] = w1[tid];
  sw2[tid] = w2[tid];
  if (tid < 16) { sb1[tid] = b1[tid]; sb2[tid] = b2[tid]; }
  __syncthreads();

  int row = blockIdx.x * 256 + tid;
  float x[16], h[16], f[16];
#pragma unroll
  for (int k = 0; k < 16; ++k) x[k] = mm[row * 16 + k];
#pragma unroll
  for (int j = 0; j < 16; ++j) {
    float a = sb1[j];
#pragma unroll
    for (int k = 0; k < 16; ++k) a += sw1[j * 16 + k] * x[k];
    h[j] = fmaxf(a, 0.0f);
  }
  float nrm2 = 0.0f;
#pragma unroll
  for (int j = 0; j < 16; ++j) {
    float a = sb2[j];
#pragma unroll
    for (int k = 0; k < 16; ++k) a += sw2[j * 16 + k] * h[k];
    f[j] = a;
    nrm2 += a * a;
  }
  float inv = 1.0f / fmaxf(sqrtf(nrm2), 1e-12f);
#pragma unroll
  for (int j = 0; j < 16; ++j) {
    float v = f[j] * inv;
    fout[row * 16 + j] = v;
    _Float16 hv = (_Float16)v;
    fhi[row * 16 + j] = hv;
    flo[row * 16 + j] = (_Float16)(v - (float)hv);
  }
}

// ---------------------------------------------------------------------------
// Kernel 3: fused streaming logsumexp of f @ f.T / T via WMMA with fp16
// hi/lo split (near-fp32 accuracy). Each wave owns a 16-row tile (A is
// loop-invariant), streams all 512 j-tiles: 2 aligned 32B loads + 2 WMMAs
// + 8 exp per lane per tile. pos/diag recomputed exactly in fp32.
// ---------------------------------------------------------------------------
__global__ void __launch_bounds__(128)
nce_kernel(const float* __restrict__ fv,
           const _Float16* __restrict__ fhi, const _Float16* __restrict__ flo,
           float* __restrict__ rowout) {
  __shared__ float posv[64], dv[64];
  int tid = threadIdx.x;
  int i0b = blockIdx.x * 64;                      // 128 blocks * 64 rows

  if (tid < 64) {
    int row  = i0b + tid;
    int prow = (row + HALF_N) & (NROWS - 1);
    const float* fr = fv + row  * 16;
    const float* fp = fv + prow * 16;
    float pos = 0.0f, d = 0.0f;
#pragma unroll
    for (int k = 0; k < 16; ++k) { pos += fr[k] * fp[k]; d += fr[k] * fr[k]; }
    posv[tid] = pos * INV_T;                      // exact positive logit
    dv[tid]   = d * INV_T;                        // exact diagonal logit
  }
  __syncthreads();

  int wave = tid >> 5;
  int lane = tid & 31;
  int i0w  = i0b + wave * 16;                     // this wave's 16 rows
  int m    = i0w + (lane & 15);
  int cb   = (lane < 16) ? 0 : 8;                 // A layout: K half per lane

  // A1 = [hi | lo] (K0-15 = hi, K16-31 = lo); A2 = [hi | 0]
  v16h ahi = *(const v16h*)(fhi + m * 16);
  v16h alo = *(const v16h*)(flo + m * 16);
  v16h a1, a2;
#pragma unroll
  for (int e = 0; e < 16; ++e) {
    a1[e] = cb ? alo[e] : ahi[e];
    a2[e] = cb ? (_Float16)0.0f : ahi[e];
  }

  float acc[8] = {0, 0, 0, 0, 0, 0, 0, 0};
  for (int j0 = 0; j0 < NROWS; j0 += 16) {
    int jr = j0 + (lane & 15);
    // B (k = 16h + s order, same as A): lane n holds K0-15, lane n+16 holds K16-31 -> row duplicated
    v16h b1v = *(const v16h*)(fhi + jr * 16);     // [Bhi | Bhi]
    v16h b2v = *(const v16h*)(flo + jr * 16);     // [Blo | Blo]
    v8f cc = {};
    cc = __builtin_amdgcn_wmma_f32_16x16x32_f16(false, a1, false, b1v,
                                                (short)0, cc, false, false);
    cc = __builtin_amdgcn_wmma_f32_16x16x32_f16(false, a2, false, b2v,
                                                (short)0, cc, false, false);
#pragma unroll
    for (int r = 0; r < 8; ++r)
      acc[r] += __expf(cc[r] * INV_T - MAXL);     // logits <= 10, no overflow
  }

  // Reduce columns across each 16-lane half (C: col = lane&15)
#pragma unroll
  for (int r = 0; r < 8; ++r) {
    float v = acc[r];
    v += __shfl_xor(v, 1, 32);
    v += __shfl_xor(v, 2, 32);
    v += __shfl_xor(v, 4, 32);
    v += __shfl_xor(v, 8, 32);
    acc[r] = v;
  }

  if ((lane & 15) == 0) {                         // lane 0 -> rows 0-7, lane 16 -> 8-15
    int mbase = i0w + ((lane >> 4) << 3);
#pragma unroll
    for (int r = 0; r < 8; ++r) {
      int row = mbase + r;
      int l   = row - i0b;
      float S   = acc[r];                         // sum over ALL j incl. diagonal
      float lse = MAXL + logf(S - __expf(dv[l] - MAXL));  // drop diagonal
      rowout[row] = lse - posv[l];
    }
  }
}

// ---------------------------------------------------------------------------
// Kernel 4: deterministic mean over the 8192 per-row losses.
// ---------------------------------------------------------------------------
__global__ void __launch_bounds__(256)
reduce_kernel(const float* __restrict__ rowout, float* __restrict__ out) {
  __shared__ float s[256];
  int tid = threadIdx.x;
  float a = 0.0f;
  for (int k = tid; k < NROWS; k += 256) a += rowout[k];
  s[tid] = a;
  __syncthreads();
  for (int off = 128; off > 0; off >>= 1) {
    if (tid < off) s[tid] += s[tid + off];
    __syncthreads();
  }
  if (tid == 0) out[0] = s[0] * (1.0f / (float)NROWS);
}

// ---------------------------------------------------------------------------
extern "C" void kernel_launch(void* const* d_in, const int* in_sizes, int n_in,
                              void* d_out, int out_size, void* d_ws, size_t ws_size,
                              hipStream_t stream) {
  const float* p1 = (const float*)d_in[0];
  const float* p2 = (const float*)d_in[1];
  const float* w1 = (const float*)d_in[2];
  const float* b1 = (const float*)d_in[3];
  const float* w2 = (const float*)d_in[4];
  const float* b2 = (const float*)d_in[5];

  char* ws = (char*)d_ws;
  float*    mm     = (float*)(ws);                 // 8192*16 f32 = 512 KB
  float*    fv     = (float*)(ws + (512u << 10));  // 8192*16 f32 = 512 KB
  _Float16* fhi    = (_Float16*)(ws + (1024u << 10)); // 256 KB
  _Float16* flo    = (_Float16*)(ws + (1280u << 10)); // 256 KB
  float*    rowout = (float*)(ws + (1536u << 10));    // 32 KB
  // total scratch: ~1.57 MB

  pool_kernel  <<<512, 256, 0, stream>>>(p1, p2, mm);
  mlp_kernel   <<< 32, 256, 0, stream>>>(mm, w1, b1, w2, b2, fv, fhi, flo);
  nce_kernel   <<<128, 128, 0, stream>>>(fv, fhi, flo, rowout);
  reduce_kernel<<<  1, 256, 0, stream>>>(rowout, (float*)d_out);
}
